// VectorQuantizeParameterize_13915694039137
// MI455X (gfx1250) — compile-verified, matrix operand order checked by reading, not run on hardware
//
#include <hip/hip_runtime.h>

typedef __attribute__((ext_vector_type(16))) __bf16 v16bf;
typedef __attribute__((ext_vector_type(8)))  __bf16 v8bf;
typedef __attribute__((ext_vector_type(8)))  float  v8f;

#define BATCH   4
#define SEQ     2048
#define BS      8192          // BATCH*SEQ rows
#define DIM     512
#define KCODES  8192
#define COMMIT_W 0.2f
#define KSTEPS  (DIM / 32)    // 16 wmma k-steps per 16x16 tile dot product

#define CHUNK_CODES 64                         // codes staged per LDS chunk
#define CHUNK_BYTES (CHUNK_CODES * DIM * 2)    // 64 KiB
#define NCHUNK      (KCODES / CHUNK_CODES)     // 128
#define ROWS_PER_BLOCK 64                      // 4 row-tiles of 16
#define NBLOCKS     (BS / ROWS_PER_BLOCK)      // 128

// ---------------- workspace layout (bytes) ----------------
#define WS_XB    0ull                                  // BS*DIM bf16   = 8 MiB
#define WS_CB    (WS_XB + (size_t)BS * DIM * 2)        // K*DIM bf16    = 8 MiB
#define WS_CNRM  (WS_CB + (size_t)KCODES * DIM * 2)    // K fp32 norms
#define WS_IND   (WS_CNRM + (size_t)KCODES * 4)        // BS int32 raw indices
#define WS_SE    (WS_IND + (size_t)BS * 4)             // fp32 masked squared error
#define WS_MS    (WS_SE + 4)                           // fp32 mask-one count
#define WS_NEED  (WS_MS + 4)

__device__ __forceinline__ unsigned short f2bf_rne(float f) {
    unsigned int u = __float_as_uint(f);
    u = (u + 0x7FFFu + ((u >> 16) & 1u)) >> 16;
    return (unsigned short)u;
}

// load a 16-element bf16 A/B fragment: 8 elems at +c0, 8 elems at +c1
__device__ __forceinline__ v16bf load_frag(const unsigned short* base, int c0, int c1) {
    v8bf lo = *(const v8bf*)(const void*)(base + c0);
    v8bf hi = *(const v8bf*)(const void*)(base + c1);
    v16bf r;
#pragma unroll
    for (int i = 0; i < 8; ++i) { r[i] = lo[i]; r[i + 8] = hi[i]; }
    return r;
}

// CDNA5 async global->LDS copy (ASYNCcnt-tracked), 16B per lane.
// VDST = 32-bit LDS byte address, ADDR = 64-bit global address.
__device__ __forceinline__ void async_b128_to_lds(unsigned lds_addr, const void* gaddr) {
    asm volatile("global_load_async_to_lds_b128 %0, %1, off"
                 :: "v"(lds_addr), "v"(gaddr) : "memory");
}
__device__ __forceinline__ void wait_asynccnt0()  { asm volatile("s_wait_asynccnt 0x0"  ::: "memory"); }
__device__ __forceinline__ void wait_asynccnt16() { asm volatile("s_wait_asynccnt 0x10" ::: "memory"); }

// -------- kernel 0: x fp32 -> bf16, and zero the scalar accumulators --------
__global__ void vq_convert_x(const float* __restrict__ x,
                             unsigned short* __restrict__ xb,
                             float* __restrict__ se, float* __restrict__ ms) {
    if (blockIdx.x == 0 && threadIdx.x == 0) { *se = 0.0f; *ms = 0.0f; }
    size_t n = (size_t)BS * DIM;
    for (size_t i = (size_t)blockIdx.x * blockDim.x + threadIdx.x; i < n;
         i += (size_t)gridDim.x * blockDim.x) {
        xb[i] = f2bf_rne(x[i]);
    }
}

// -------- kernel 1: codebook fp32 -> bf16 + squared norms (block per code) --------
__global__ void vq_cb_prep(const float* __restrict__ cb,
                           unsigned short* __restrict__ cbb,
                           float* __restrict__ cnorm) {
    const int k   = blockIdx.x;
    const int tid = threadIdx.x;
    const float* row = cb + (size_t)k * DIM;
    unsigned short* rowb = cbb + (size_t)k * DIM;
    float acc = 0.0f;
#pragma unroll
    for (int d = tid; d < DIM; d += 256) {
        float v = row[d];
        rowb[d] = f2bf_rne(v);
        acc += v * v;
    }
#pragma unroll
    for (int off = 16; off > 0; off >>= 1) acc += __shfl_down(acc, off, 32);
    __shared__ float ssum[8];
    if ((tid & 31) == 0) ssum[tid >> 5] = acc;
    __syncthreads();
    if (tid == 0) {
        float t = 0.0f;
#pragma unroll
        for (int w = 0; w < 8; ++w) t += ssum[w];
        cnorm[k] = t;
    }
}

// -------- kernel 2: fused bf16-WMMA GEMM + row argmin, LDS-staged codebook --------
// 128 blocks x 256 threads (8 waves). Block owns 64 rows (4 M-tiles).
// Wave w: row-tile = w&3, code-half = w>>2 (32 of the 64 staged codes).
// Codebook chunks double-buffered in LDS via global_load_async_to_lds_b128;
// B fragments software-pipelined in registers so ds_load overlaps v_wmma.
__global__ __launch_bounds__(256)
void vq_argmin(const unsigned short* __restrict__ xb,
               const unsigned short* __restrict__ cbb,
               const float* __restrict__ cnorm,
               const int* __restrict__ mask,
               int* __restrict__ ind_raw,
               float* __restrict__ out_idx_f) {
    __shared__ unsigned short sB[2][CHUNK_CODES * DIM];   // 2 x 64 KiB
    __shared__ float md[2][4][16];
    __shared__ int   mi[2][4][16];

    const int tid   = threadIdx.x;
    const int wave  = tid >> 5;
    const int lane  = tid & 31;
    const int l16   = lane & 15;
    const int half  = lane >> 4;          // K sub-chunk select (ISA 16-bit A/B layout)
    const int rt    = wave & 3;           // row tile within block
    const int chalf = wave >> 2;          // code half within staged chunk
    const int rbase = blockIdx.x * ROWS_PER_BLOCK;

    // ---- preload this wave's A tile (16 rows x 512 K, bf16) into registers ----
    const unsigned short* arow = xb + (size_t)(rbase + rt * 16 + l16) * DIM;
    v16bf afrag[KSTEPS];
#pragma unroll
    for (int ks = 0; ks < KSTEPS; ++ks) {
        const int k0 = ks * 32;
        afrag[ks] = load_frag(arow, k0 + half * 8, k0 + 16 + half * 8);
    }

    float best[8];
    int   bidx[8];
#pragma unroll
    for (int m = 0; m < 8; ++m) { best[m] = 3.0e38f; bidx[m] = 0; }

    const unsigned lds0 = (unsigned)(size_t)&sB[0][0];
    const unsigned lds1 = (unsigned)(size_t)&sB[1][0];

    // stage chunk c into LDS buffer: 16 async b128 per thread
#define STAGE(c, ldsbase)                                                     \
    {                                                                         \
        const char* g = (const char*)cbb + (size_t)(c) * CHUNK_BYTES;         \
        _Pragma("unroll")                                                     \
        for (int it = 0; it < 16; ++it) {                                     \
            const int o = (it * 256 + tid) * 16;                              \
            async_b128_to_lds((ldsbase) + o, g + o);                          \
        }                                                                     \
    }

    STAGE(0, lds0)
    for (int c = 0; c < NCHUNK; ++c) {
        const unsigned short* buf = &sB[c & 1][0];
        if (c + 1 < NCHUNK) {
            STAGE(c + 1, ((c + 1) & 1) ? lds1 : lds0)
            wait_asynccnt16();            // chunk c's 16 asyncs done (in-order)
        } else {
            wait_asynccnt0();
        }
        __syncthreads();                  // all waves' staging visible

        // two 16-code tiles, processed together with pipelined B fragments
        const unsigned short* brow0 = buf + (chalf * 32 + l16) * DIM;
        const unsigned short* brow1 = brow0 + 16 * DIM;
        v8f acc0 = {}; v8f acc1 = {};
        v16bf b0 = load_frag(brow0, half * 8, 16 + half * 8);
        v16bf b1 = load_frag(brow1, half * 8, 16 + half * 8);
#pragma unroll
        for (int ks = 0; ks < KSTEPS; ++ks) {
            v16bf nb0, nb1;
            if (ks + 1 < KSTEPS) {
                const int k1 = (ks + 1) * 32;
                nb0 = load_frag(brow0, k1 + half * 8, k1 + 16 + half * 8);
                nb1 = load_frag(brow1, k1 + half * 8, k1 + 16 + half * 8);
            }
            acc0 = __builtin_amdgcn_wmma_f32_16x16x32_bf16(
                       false, afrag[ks], false, b0, (short)0, acc0, false, false);
            acc1 = __builtin_amdgcn_wmma_f32_16x16x32_bf16(
                       false, afrag[ks], false, b1, (short)0, acc1, false, false);
            if (ks + 1 < KSTEPS) { b0 = nb0; b1 = nb1; }
        }

        // distance surrogate: ||c||^2 - 2 x.c  (||x||^2 is row-constant)
        const int n0 = c * CHUNK_CODES + chalf * 32 + l16;
        const int n1 = n0 + 16;
        const float cn0 = cnorm[n0];
        const float cn1 = cnorm[n1];
#pragma unroll
        for (int m = 0; m < 8; ++m) {
            float s0 = __builtin_fmaf(-2.0f, acc0[m], cn0);
            float s1 = __builtin_fmaf(-2.0f, acc1[m], cn1);
            if (s0 < best[m] || (s0 == best[m] && n0 < bidx[m])) { best[m] = s0; bidx[m] = n0; }
            if (s1 < best[m] || (s1 == best[m] && n1 < bidx[m])) { best[m] = s1; bidx[m] = n1; }
        }
        __syncthreads();                  // done reading buf before it is refilled
    }
#undef STAGE

    // butterfly min-reduce over the 16 lanes of each half (C layout: N = lane&15)
#pragma unroll
    for (int off = 8; off > 0; off >>= 1) {
#pragma unroll
        for (int m = 0; m < 8; ++m) {
            float od = __shfl_xor(best[m], off, 16);
            int   oi = __shfl_xor(bidx[m], off, 16);
            if (od < best[m] || (od == best[m] && oi < bidx[m])) { best[m] = od; bidx[m] = oi; }
        }
    }

    // C layout: lanes 0-15 hold rows M = vgpr, lanes 16-31 hold rows M = vgpr+8
    if (lane == 0) {
#pragma unroll
        for (int m = 0; m < 8; ++m) { md[chalf][rt][m] = best[m]; mi[chalf][rt][m] = bidx[m]; }
    } else if (lane == 16) {
#pragma unroll
        for (int m = 0; m < 8; ++m) { md[chalf][rt][8 + m] = best[m]; mi[chalf][rt][8 + m] = bidx[m]; }
    }
    __syncthreads();

    // merge the two code-half waves per row, write results
    if (tid < 64) {
        const int r = tid >> 4, m = tid & 15;
        float d0 = md[0][r][m]; int i0 = mi[0][r][m];
        float d1 = md[1][r][m]; int i1 = mi[1][r][m];
        if (d1 < d0 || (d1 == d0 && i1 < i0)) { d0 = d1; i0 = i1; }
        const int row = rbase + r * 16 + m;
        ind_raw[row]   = i0;
        out_idx_f[row] = (float)(i0 * mask[row]);
    }
}

// -------- kernel 3: gather + masked output + masked SE / mask count --------
__global__ __launch_bounds__(256)
void vq_output(const float* __restrict__ x,
               const float* __restrict__ cb,
               const int* __restrict__ mask,
               const int* __restrict__ ind_raw,
               float* __restrict__ out_q,       // d_out + BS
               float* __restrict__ se_acc,
               float* __restrict__ ms_acc) {
    const int row  = blockIdx.x;
    const int tid  = threadIdx.x;
    const int m    = mask[row];
    const float mf = (float)m;
    const int ind  = ind_raw[row];
    const float* xr = x  + (size_t)row * DIM;
    const float* cr = cb + (size_t)ind * DIM;
    float* orow = out_q + (size_t)row * DIM;

    float se = 0.0f;
#pragma unroll
    for (int d = tid; d < DIM; d += 256) {
        float q  = cr[d];
        float xv = xr[d];
        orow[d] = q * mf;                  // straight-through value * mask
        float e = q - xv;
        se = __builtin_fmaf(e * mf, e, se);
    }
#pragma unroll
    for (int off = 16; off > 0; off >>= 1) se += __shfl_down(se, off, 32);
    __shared__ float ssum[8];
    if ((tid & 31) == 0) ssum[tid >> 5] = se;
    __syncthreads();
    if (tid == 0) {
        float t = 0.0f;
#pragma unroll
        for (int w = 0; w < 8; ++w) t += ssum[w];
        atomicAdd(se_acc, t);
        atomicAdd(ms_acc, mf);
    }
}

// -------- kernel 4: scalar loss --------
__global__ void vq_finalize(const float* __restrict__ se_acc,
                            const float* __restrict__ ms_acc,
                            float* __restrict__ loss_out) {
    float se = *se_acc;
    float ms = *ms_acc;
    float denom = fmaxf(ms * (float)DIM, 1.0f);
    *loss_out = (se / denom) * COMMIT_W * ms;
}

extern "C" void kernel_launch(void* const* d_in, const int* in_sizes, int n_in,
                              void* d_out, int out_size, void* d_ws, size_t ws_size,
                              hipStream_t stream) {
    (void)in_sizes; (void)n_in; (void)out_size;
    if (ws_size < WS_NEED) return;

    const float* x   = (const float*)d_in[0];
    const float* cb  = (const float*)d_in[1];
    const int*   msk = (const int*)d_in[2];
    float* out = (float*)d_out;          // [BS idx][BS*DIM output][1 loss]

    char* ws = (char*)d_ws;
    unsigned short* xb    = (unsigned short*)(ws + WS_XB);
    unsigned short* cbb   = (unsigned short*)(ws + WS_CB);
    float*          cnorm = (float*)(ws + WS_CNRM);
    int*            ind   = (int*)(ws + WS_IND);
    float*          se    = (float*)(ws + WS_SE);
    float*          ms    = (float*)(ws + WS_MS);

    vq_convert_x<<<1024, 256, 0, stream>>>(x, xb, se, ms);
    vq_cb_prep<<<KCODES, 256, 0, stream>>>(cb, cbb, cnorm);
    vq_argmin<<<NBLOCKS, 256, 0, stream>>>(xb, cbb, cnorm, msk, ind, out);
    vq_output<<<BS, 256, 0, stream>>>(x, cb, msk, ind, out + BS, se, ms);
    vq_finalize<<<1, 1, 0, stream>>>(se, ms, out + (size_t)BS + (size_t)BS * DIM);
}
